// RQBottleneckTransformer_30571577213324
// MI455X (gfx1250) — compile-verified, matrix operand order checked by reading, not run on hardware
//
#include <hip/hip_runtime.h>
#include <hip/hip_bf16.h>

typedef __attribute__((ext_vector_type(16))) _Float16 v16h;
typedef __attribute__((ext_vector_type(8)))  _Float16 v8h;
typedef __attribute__((ext_vector_type(8)))  float    v8f;
typedef __attribute__((ext_vector_type(4)))  float    v4f;

// ---------------------------------------------------------------- constants
#define BB 8
#define TT 1500
#define DD 512
#define HH 8
#define HDD 64
#define TS 750
#define FFD 2048
#define CBN 1025
#define CDD 64
#define XTOK (BB*TS)      // 6000 downsampled tokens
#define FTOK (BB*TT)      // 12000 full tokens
#define TP 1504           // padded T for V^T (16B-aligned f16 rows)

__device__ __forceinline__ int kpair_base(int p, int hi) {
  return ((p < 4) ? (p * 2) : (16 + (p - 4) * 2)) + hi;
}

__device__ __forceinline__ float gelu_tanh(float x) {
  const float c = 0.7978845608028654f; // sqrt(2/pi)
  float x3 = x * x * x;
  return 0.5f * x * (1.0f + tanhf(c * (x + 0.044715f * x3)));
}

union F16X16 { v16h v; v8h h8[2]; };

// Load a 16-half WMMA fragment: two contiguous 8-half (16B) segments at
// base[0..7] (K = hi..hi+7) and base[16..23] (K = 16+hi..23+hi).
__device__ __forceinline__ v16h load_frag16(const _Float16* base) {
  F16X16 f;
  f.h8[0] = *(const v8h*)(base);
  f.h8[1] = *(const v8h*)(base + 16);
  return f.v;
}

#define WMMA_F16(a, b, c) \
  __builtin_amdgcn_wmma_f32_16x16x32_f16(false, (a), false, (b), (short)0, (c), false, false)

// ---------------------------------------------------------------- WMMA GEMM
// C[M,N] = epi(A[M,K](f16) @ BT[N,K](f16)); flags: 1 bias, 2 gelu, 4 resid,
// 8 = f16 output (Ch), else f32 output (Cf). Register/L2 streaming, no LDS.
// Each wave: 32 rows x 64 cols = 2 A-frags sharing 4 B-frags -> 8 WMMA/k-step.
#define GBM 256   // 8 waves x 32 rows
#define GBN 64

__global__ __launch_bounds__(256) void gemm_wmma_kernel(
    const _Float16* __restrict__ A,  const _Float16* __restrict__ BT,
    const float* __restrict__ bias,  const float* __restrict__ resid,
    float* __restrict__ Cf, _Float16* __restrict__ Ch,
    int M, int N, int K, int flags)
{
  const int lane = threadIdx.x & 31;
  const int w    = threadIdx.x >> 5;        // 0..7
  const int l15  = lane & 15;
  const int hi   = (lane >> 4) * 8;
  const int m0   = blockIdx.y * GBM + w * 32;
  const int n0   = blockIdx.x * GBN;

  int arow0 = m0 + l15;      if (arow0 > M - 1) arow0 = M - 1;
  int arow1 = m0 + 16 + l15; if (arow1 > M - 1) arow1 = M - 1;
  const _Float16* aptr0 = A + (size_t)arow0 * K + hi;
  const _Float16* aptr1 = A + (size_t)arow1 * K + hi;
  const _Float16* bptr  = BT + (size_t)(n0 + l15) * K + hi;

  v8f acc0[4] = {{}, {}, {}, {}};
  v8f acc1[4] = {{}, {}, {}, {}};

  #pragma unroll 2
  for (int kt = 0; kt < K; kt += 32) {
    if (kt + 64 < K) {
      __builtin_prefetch(aptr0 + kt + 64, 0, 1);
      __builtin_prefetch(aptr1 + kt + 64, 0, 1);
    }
    // batch all loads first: 2 A frags + 4 B frags (12x b128 in one clause)
    v16h a0 = load_frag16(aptr0 + kt);
    v16h a1 = load_frag16(aptr1 + kt);
    v16h b0 = load_frag16(bptr + (size_t)( 0) * K + kt);
    v16h b1 = load_frag16(bptr + (size_t)(16) * K + kt);
    v16h b2 = load_frag16(bptr + (size_t)(32) * K + kt);
    v16h b3 = load_frag16(bptr + (size_t)(48) * K + kt);
    acc0[0] = WMMA_F16(a0, b0, acc0[0]);
    acc0[1] = WMMA_F16(a0, b1, acc0[1]);
    acc0[2] = WMMA_F16(a0, b2, acc0[2]);
    acc0[3] = WMMA_F16(a0, b3, acc0[3]);
    acc1[0] = WMMA_F16(a1, b0, acc1[0]);
    acc1[1] = WMMA_F16(a1, b1, acc1[1]);
    acc1[2] = WMMA_F16(a1, b2, acc1[2]);
    acc1[3] = WMMA_F16(a1, b3, acc1[3]);
  }

  // epilogue: C layout row = v + hi, col = l15 (per 16-wide subtile)
  #pragma unroll
  for (int half = 0; half < 2; ++half) {
    v8f* acc = half ? acc1 : acc0;
    #pragma unroll
    for (int j = 0; j < 4; ++j) {
      int col = n0 + 16 * j + l15;
      #pragma unroll
      for (int v = 0; v < 8; ++v) {
        int row = m0 + half * 16 + v + hi;
        if (row < M) {
          float r = acc[j][v];
          if (flags & 1) r += bias[col];
          if (flags & 2) r = gelu_tanh(r);
          if (flags & 4) r += resid[(size_t)row * N + col];
          if (flags & 8) Ch[(size_t)row * N + col] = (_Float16)r;
          else           Cf[(size_t)row * N + col] = r;
        }
      }
    }
  }
}

// ---------------------------------------------------------------- attention
// 8 waves/block, one 16-query tile per wave, flash-style online softmax.
__global__ __launch_bounds__(256) void attn_wmma_kernel(
    const _Float16* __restrict__ Q, const _Float16* __restrict__ Kh,
    const _Float16* __restrict__ VT, _Float16* __restrict__ O)
{
  __shared__ _Float16 p_lds[8][16][34];

  const int lane = threadIdx.x & 31;
  const int w    = threadIdx.x >> 5;
  const int l15  = lane & 15;
  const int hi   = (lane >> 4) * 8;
  const int h    = blockIdx.y;
  const int b    = blockIdx.z;
  const int q0   = (blockIdx.x * 8 + w) * 16;

  int qr = q0 + l15; if (qr > TT - 1) qr = TT - 1;
  const _Float16* qrow = Q + ((size_t)b * TT + qr) * DD + h * HDD + hi;
  const v16h aq0 = load_frag16(qrow);        // d = 0..31 chunk
  const v16h aq1 = load_frag16(qrow + 32);   // d = 32..63 chunk

  v8f oacc[4] = {{}, {}, {}, {}};
  float m_i[8], l_i[8];
  #pragma unroll
  for (int v = 0; v < 8; ++v) { m_i[v] = -1e30f; l_i[v] = 0.0f; }

  const _Float16* vbase = VT + ((size_t)b * HH + h) * HDD * (size_t)TP + hi;

  const int nkt = (TT + 31) / 32;  // 47
  for (int kt = 0; kt < nkt; ++kt) {
    const int k0 = kt * 32;

    // S = Q @ K^T : preload all 4 K frags, then 4 WMMAs
    int key0 = k0 + l15;      if (key0 > TT - 1) key0 = TT - 1;
    int key1 = k0 + 16 + l15; if (key1 > TT - 1) key1 = TT - 1;
    const _Float16* kr0 = Kh + ((size_t)b * TT + key0) * DD + h * HDD + hi;
    const _Float16* kr1 = Kh + ((size_t)b * TT + key1) * DD + h * HDD + hi;
    v16h bk00 = load_frag16(kr0);
    v16h bk01 = load_frag16(kr0 + 32);
    v16h bk10 = load_frag16(kr1);
    v16h bk11 = load_frag16(kr1 + 32);
    v8f s0 = {}, s1 = {};
    s0 = WMMA_F16(aq0, bk00, s0);
    s0 = WMMA_F16(aq1, bk01, s0);
    s1 = WMMA_F16(aq0, bk10, s1);
    s1 = WMMA_F16(aq1, bk11, s1);

    // issue V fragment loads now so they hide under the softmax VALU work
    v16h bv0 = load_frag16(vbase + (size_t)( 0 + l15) * TP + k0);
    v16h bv1 = load_frag16(vbase + (size_t)(16 + l15) * TP + k0);
    v16h bv2 = load_frag16(vbase + (size_t)(32 + l15) * TP + k0);
    v16h bv3 = load_frag16(vbase + (size_t)(48 + l15) * TP + k0);

    // mask tail keys (QK_SCALE = 8/sqrt(64) = 1.0 exactly)
    const bool bad0 = (k0 + l15) >= TT;
    const bool bad1 = (k0 + 16 + l15) >= TT;
    #pragma unroll
    for (int v = 0; v < 8; ++v) {
      if (bad0) s0[v] = -1e30f;
      if (bad1) s1[v] = -1e30f;
    }

    // online softmax: a row's 32 cols live across one 16-lane half
    #pragma unroll
    for (int v = 0; v < 8; ++v) {
      float mx = fmaxf(s0[v], s1[v]);
      #pragma unroll
      for (int off = 8; off >= 1; off >>= 1)
        mx = fmaxf(mx, __shfl_xor(mx, off, 32));
      float mnew  = fmaxf(m_i[v], mx);
      float alpha = __expf(m_i[v] - mnew);
      float p0 = __expf(s0[v] - mnew);
      float p1 = __expf(s1[v] - mnew);
      float ps = p0 + p1;
      #pragma unroll
      for (int off = 8; off >= 1; off >>= 1)
        ps += __shfl_xor(ps, off, 32);
      l_i[v] = l_i[v] * alpha + ps;
      m_i[v] = mnew;
      #pragma unroll
      for (int j = 0; j < 4; ++j) oacc[j][v] *= alpha;
      s0[v] = p0; s1[v] = p1;
    }

    // P (C-layout) -> LDS -> A-layout fragment
    #pragma unroll
    for (int v = 0; v < 8; ++v) {
      p_lds[w][v + hi][l15]      = (_Float16)s0[v];
      p_lds[w][v + hi][16 + l15] = (_Float16)s1[v];
    }
    __builtin_amdgcn_wave_barrier();
    v16h ap;
    #pragma unroll
    for (int p = 0; p < 8; ++p) {
      int kb = kpair_base(p, hi);
      ap[2*p]   = p_lds[w][l15][kb];
      ap[2*p+1] = p_lds[w][l15][kb + 1];
    }
    __builtin_amdgcn_wave_barrier();

    // O += P @ V (V^T zero-padded: masked keys contribute exact 0)
    oacc[0] = WMMA_F16(ap, bv0, oacc[0]);
    oacc[1] = WMMA_F16(ap, bv1, oacc[1]);
    oacc[2] = WMMA_F16(ap, bv2, oacc[2]);
    oacc[3] = WMMA_F16(ap, bv3, oacc[3]);
  }

  if (q0 < TT) {
    #pragma unroll
    for (int v = 0; v < 8; ++v) {
      int row = q0 + v + hi;
      if (row < TT) {
        float inv = 1.0f / l_i[v];
        _Float16* orow = O + ((size_t)b * TT + row) * DD + h * HDD;
        #pragma unroll
        for (int j = 0; j < 4; ++j)
          orow[j * 16 + l15] = (_Float16)(oacc[j][v] * inv);
      }
    }
  }
}

// ---------------------------------------------------------------- layernorm
template <typename OT>
__global__ __launch_bounds__(256) void layernorm_kernel(
    const float* __restrict__ X, const float* __restrict__ g,
    const float* __restrict__ be, OT* __restrict__ Y)
{
  __shared__ float s_sum[256], s_sq[256];
  const int row = blockIdx.x;
  const float* x = X + (size_t)row * DD;
  float v0 = x[threadIdx.x];
  float v1 = x[threadIdx.x + 256];
  s_sum[threadIdx.x] = v0 + v1;
  s_sq[threadIdx.x]  = v0 * v0 + v1 * v1;
  __syncthreads();
  for (int off = 128; off >= 1; off >>= 1) {
    if (threadIdx.x < off) {
      s_sum[threadIdx.x] += s_sum[threadIdx.x + off];
      s_sq[threadIdx.x]  += s_sq[threadIdx.x + off];
    }
    __syncthreads();
  }
  float mean = s_sum[0] * (1.0f / DD);
  float var  = s_sq[0] * (1.0f / DD) - mean * mean;
  float inv  = rsqrtf(var + 1e-5f);
  OT* y = Y + (size_t)row * DD;
  int d0 = threadIdx.x, d1 = threadIdx.x + 256;
  y[d0] = (OT)((v0 - mean) * inv * g[d0] + be[d0]);
  y[d1] = (OT)((v1 - mean) * inv * g[d1] + be[d1]);
}

// ---------------------------------------------------------------- converts
__global__ __launch_bounds__(256) void f32_to_f16_kernel(
    const float* __restrict__ in, _Float16* __restrict__ out, int n8)
{
  int i = blockIdx.x * 256 + threadIdx.x;
  if (i < n8) {
    const v4f* p = (const v4f*)in + (size_t)i * 2;
    v4f a = p[0], bq = p[1];
    v8h hh;
    hh[0] = (_Float16)a[0];  hh[1] = (_Float16)a[1];
    hh[2] = (_Float16)a[2];  hh[3] = (_Float16)a[3];
    hh[4] = (_Float16)bq[0]; hh[5] = (_Float16)bq[1];
    hh[6] = (_Float16)bq[2]; hh[7] = (_Float16)bq[3];
    *((v8h*)out + i) = hh;
  }
}

// weight [K][N] f32 -> [N][K] f16, 32x32 LDS tiles
__global__ __launch_bounds__(256) void wtrans_kernel(
    const float* __restrict__ Wkn, _Float16* __restrict__ WT, int K, int N)
{
  __shared__ float tile[32][33];
  int k0 = blockIdx.y * 32, n0 = blockIdx.x * 32;
  #pragma unroll
  for (int i = 0; i < 4; ++i) {
    int idx = threadIdx.x + i * 256;
    int r = idx >> 5, c = idx & 31;
    tile[r][c] = Wkn[(size_t)(k0 + r) * N + n0 + c];
  }
  __syncthreads();
  #pragma unroll
  for (int i = 0; i < 4; ++i) {
    int idx = threadIdx.x + i * 256;
    int r = idx >> 5, c = idx & 31;
    WT[(size_t)(n0 + r) * K + k0 + c] = (_Float16)tile[c][r];
  }
}

// V f16 [B][T][D] -> V^T f16 [B][H][HD][TP], zero-padded tail
__global__ __launch_bounds__(256) void vtrans_kernel(
    const _Float16* __restrict__ V, _Float16* __restrict__ VT)
{
  __shared__ _Float16 tile[32][HDD + 4];
  int b = blockIdx.z, h = blockIdx.y, t0 = blockIdx.x * 32;
  #pragma unroll
  for (int i = 0; i < 8; ++i) {
    int idx = threadIdx.x + i * 256;
    int t = idx >> 6, d = idx & 63;
    int gt = t0 + t;
    _Float16 val = (_Float16)0.0f;
    if (gt < TT) val = V[((size_t)b * TT + gt) * DD + h * HDD + d];
    tile[t][d] = val;
  }
  __syncthreads();
  #pragma unroll
  for (int i = 0; i < 8; ++i) {
    int idx = threadIdx.x + i * 256;
    int d = idx >> 5, t = idx & 31;
    VT[(((size_t)b * HH + h) * HDD + d) * TP + t0 + t] = tile[t][d];
  }
}

// ---------------------------------------------------------------- small ops
__global__ void downsample_kernel(const float* __restrict__ embs,
                                  float* __restrict__ x0)
{
  int r = blockIdx.x;                 // b*TS + ts
  int b = r / TS, ts = r % TS;
  const float* src = embs + ((size_t)b * TT + (size_t)ts * 2) * DD;
  float* dst = x0 + (size_t)r * DD;
  dst[threadIdx.x]       = src[threadIdx.x];
  dst[threadIdx.x + 256] = src[threadIdx.x + 256];
}

__global__ __launch_bounds__(256) void codebook_kernel(
    const float* __restrict__ xq, const float* __restrict__ cb,
    _Float16* __restrict__ qv, float* __restrict__ loss)
{
  __shared__ float sx[8][CDD];
  const int w = threadIdx.x >> 5, lane = threadIdx.x & 31;
  const int token = blockIdx.x * 8 + w;
  const float* xr = xq + (size_t)token * CDD;
  sx[w][lane]      = xr[lane];
  sx[w][lane + 32] = xr[lane + 32];
  __syncthreads();

  float best = 3.4e38f; int bidx = 0;
  for (int c = lane; c < CBN; c += 32) {
    const float* cr = cb + (size_t)c * CDD;
    float d2 = 0.0f;
    #pragma unroll 8
    for (int d = 0; d < CDD; ++d) {
      float df = sx[w][d] - cr[d];
      d2 = fmaf(df, df, d2);
    }
    if (d2 < best || (d2 == best && c < bidx)) { best = d2; bidx = c; }
  }
  #pragma unroll
  for (int off = 16; off >= 1; off >>= 1) {
    float ob = __shfl_xor(best, off, 32);
    int   oi = __shfl_xor(bidx, off, 32);
    if (ob < best || (ob == best && oi < bidx)) { best = ob; bidx = oi; }
  }

  const float* cr = cb + (size_t)bidx * CDD;
  float q0 = cr[lane], q1 = cr[lane + 32];
  qv[(size_t)token * CDD + lane]      = (_Float16)q0;
  qv[(size_t)token * CDD + lane + 32] = (_Float16)q1;
  float d0 = q0 - sx[w][lane], d1 = q1 - sx[w][lane + 32];
  float part = d0 * d0 + d1 * d1;
  #pragma unroll
  for (int off = 16; off >= 1; off >>= 1) part += __shfl_xor(part, off, 32);
  if (lane == 0) atomicAdd(loss, part);
}

__global__ void maskvec_kernel(const float* __restrict__ cb,
                               const float* __restrict__ pw,
                               const float* __restrict__ pb,
                               float* __restrict__ mv)
{
  int d = blockIdx.x * 256 + threadIdx.x;
  const float* c = cb + (size_t)(CBN - 1) * CDD;
  float acc = pb[d];
  #pragma unroll 8
  for (int k = 0; k < CDD; ++k) acc = fmaf(c[k], pw[(size_t)k * DD + d], acc);
  mv[d] = acc;
}

__global__ void expand_kernel(const float* __restrict__ xo,
                              const unsigned char* __restrict__ mask,
                              const float* __restrict__ mv,
                              const float* __restrict__ pos,
                              float* __restrict__ X)
{
  int r = blockIdx.x;                 // b*TT + t
  int b = r / TT, t = r % TT;
  bool mk = mask[r] != 0;
  const float* src = xo + ((size_t)b * TS + (t >> 1)) * DD;
  const float* pp  = pos + (size_t)t * DD;
  float* dst = X + (size_t)r * DD;
  #pragma unroll
  for (int i = 0; i < 2; ++i) {
    int d = threadIdx.x + i * 256;
    float v = mk ? src[d] : mv[d];
    dst[d] = v + pp[d];
  }
}

__global__ void loss_final_kernel(const float* __restrict__ acc,
                                  float* __restrict__ out)
{
  out[0] = acc[0] / (float)(XTOK * CDD);
}

// ---------------------------------------------------------------- launcher
extern "C" void kernel_launch(void* const* d_in, const int* in_sizes, int n_in,
                              void* d_out, int out_size, void* d_ws, size_t ws_size,
                              hipStream_t stream) {
  (void)in_sizes; (void)n_in; (void)out_size; (void)ws_size;

  const float* embs         = (const float*)d_in[0];
  const unsigned char* mask = (const unsigned char*)d_in[1];
  const float* mlp_ln_g   = (const float*)d_in[2];
  const float* mlp_ln_b   = (const float*)d_in[3];
  const float* mlp_w1     = (const float*)d_in[4];
  const float* mlp_b1     = (const float*)d_in[5];
  const float* mlp_w2     = (const float*)d_in[6];
  const float* mlp_b2     = (const float*)d_in[7];
  const float* proj_in_w  = (const float*)d_in[8];
  const float* proj_in_b  = (const float*)d_in[9];
  const float* codebook   = (const float*)d_in[10];
  const float* proj_out_w = (const float*)d_in[11];
  const float* proj_out_b = (const float*)d_in[12];
  const float* pos_emb    = (const float*)d_in[13];
  const float* attn_ln_g  = (const float*)d_in[14];
  const float* attn_ln_b  = (const float*)d_in[15];
  const float* wq         = (const float*)d_in[16];
  const float* bq         = (const float*)d_in[17];
  const float* wk         = (const float*)d_in[18];
  const float* bk         = (const float*)d_in[19];
  const float* wv         = (const float*)d_in[20];
  const float* bv         = (const float*)d_in[21];
  const float* wo         = (const float*)d_in[22];
  const float* bo         = (const float*)d_in[23];
  const float* ffn_ln_g   = (const float*)d_in[24];
  const float* ffn_ln_b   = (const float*)d_in[25];
  const float* ffn_w1     = (const float*)d_in[26];
  const float* ffn_b1     = (const float*)d_in[27];
  const float* ffn_w2     = (const float*)d_in[28];
  const float* ffn_b2     = (const float*)d_in[29];
  const float* ln_post_g  = (const float*)d_in[30];
  const float* ln_post_b  = (const float*)d_in[31];

  float* out = (float*)d_out;

  // ---- workspace layout: f32 region then f16 region
  float* f32p = (float*)d_ws;
  float* buf_x   = f32p;  f32p += (size_t)FTOK * DD;   // residual stream
  float* buf_x0  = f32p;  f32p += (size_t)XTOK * DD;   // downsampled stream
  float* buf_xo  = f32p;  f32p += (size_t)XTOK * DD;   // proj_out result
  float* buf_xq  = f32p;  f32p += (size_t)XTOK * CDD;
  float* buf_mv  = f32p;  f32p += DD;
  float* buf_loss= f32p;  f32p += 1;

  _Float16* h16p = (_Float16*)f32p;
  _Float16* h_ln = h16p;  h16p += (size_t)FTOK * DD;
  _Float16* h_ff = h16p;  h16p += (size_t)FTOK * FFD;
  _Float16* h_x0 = h16p;  h16p += (size_t)XTOK * DD;
  _Float16* h_qv = h16p;  h16p += (size_t)XTOK * CDD;
  _Float16* h_q  = h16p;  h16p += (size_t)FTOK * DD;
  _Float16* h_k  = h16p;  h16p += (size_t)FTOK * DD;
  _Float16* h_v  = h16p;  h16p += (size_t)FTOK * DD;
  _Float16* h_vt = h16p;  h16p += (size_t)BB * HH * HDD * TP;
  _Float16* h_o  = h16p;  h16p += (size_t)FTOK * DD;
  _Float16* wt_mlp1 = h16p; h16p += (size_t)DD * FFD;
  _Float16* wt_mlp2 = h16p; h16p += (size_t)FFD * DD;
  _Float16* wt_pin  = h16p; h16p += (size_t)DD * CDD;
  _Float16* wt_pout = h16p; h16p += (size_t)CDD * DD;
  _Float16* wt_q    = h16p; h16p += (size_t)DD * DD;
  _Float16* wt_k    = h16p; h16p += (size_t)DD * DD;
  _Float16* wt_v    = h16p; h16p += (size_t)DD * DD;
  _Float16* wt_o    = h16p; h16p += (size_t)DD * DD;
  _Float16* wt_ff1  = h16p; h16p += (size_t)DD * FFD;
  _Float16* wt_ff2  = h16p; h16p += (size_t)FFD * DD;

  auto gemm = [&](const _Float16* A, const _Float16* BT, const float* bias,
                  const float* res, float* Cf, _Float16* Ch,
                  int M, int N, int K, int flags) {
    dim3 g(N / GBN, (M + GBM - 1) / GBM);
    gemm_wmma_kernel<<<g, 256, 0, stream>>>(A, BT, bias, res, Cf, Ch, M, N, K, flags);
  };
  auto wtrans = [&](const float* Wkn, _Float16* WT, int K, int N) {
    wtrans_kernel<<<dim3(N / 32, K / 32), 256, 0, stream>>>(Wkn, WT, K, N);
  };

  hipMemsetAsync(buf_loss, 0, sizeof(float), stream);

  // pre-transpose all weights to f16 [N][K]
  wtrans(mlp_w1, wt_mlp1, DD, FFD);
  wtrans(mlp_w2, wt_mlp2, FFD, DD);
  wtrans(proj_in_w, wt_pin, DD, CDD);
  wtrans(proj_out_w, wt_pout, CDD, DD);
  wtrans(wq, wt_q, DD, DD);
  wtrans(wk, wt_k, DD, DD);
  wtrans(wv, wt_v, DD, DD);
  wtrans(wo, wt_o, DD, DD);
  wtrans(ffn_w1, wt_ff1, DD, FFD);
  wtrans(ffn_w2, wt_ff2, FFD, DD);

  // x = embs[:, ::2]
  downsample_kernel<<<XTOK, 256, 0, stream>>>(embs, buf_x0);

  // MLP block (downsampled tokens)
  layernorm_kernel<_Float16><<<XTOK, 256, 0, stream>>>(buf_x0, mlp_ln_g, mlp_ln_b, h_ln);
  gemm(h_ln, wt_mlp1, mlp_b1, nullptr, nullptr, h_ff, XTOK, FFD, DD, 1 | 2 | 8);
  gemm(h_ff, wt_mlp2, mlp_b2, buf_x0, buf_x0, nullptr, XTOK, DD, FFD, 1 | 4);

  // VQ bottleneck
  {
    int n8 = (XTOK * DD) / 8;
    f32_to_f16_kernel<<<(n8 + 255) / 256, 256, 0, stream>>>(buf_x0, h_x0, n8);
  }
  gemm(h_x0, wt_pin, proj_in_b, nullptr, buf_xq, nullptr, XTOK, CDD, DD, 1);
  codebook_kernel<<<XTOK / 8, 256, 0, stream>>>(buf_xq, codebook, h_qv, buf_loss);
  maskvec_kernel<<<DD / 256, 256, 0, stream>>>(codebook, proj_out_w, proj_out_b, buf_mv);
  gemm(h_qv, wt_pout, proj_out_b, nullptr, buf_xo, nullptr, XTOK, DD, CDD, 1);

  // repeat x2, mask, +pos_emb
  expand_kernel<<<FTOK, 256, 0, stream>>>(buf_xo, mask, buf_mv, pos_emb, buf_x);

  // attention block
  layernorm_kernel<_Float16><<<FTOK, 256, 0, stream>>>(buf_x, attn_ln_g, attn_ln_b, h_ln);
  gemm(h_ln, wt_q, bq, nullptr, nullptr, h_q, FTOK, DD, DD, 1 | 8);
  gemm(h_ln, wt_k, bk, nullptr, nullptr, h_k, FTOK, DD, DD, 1 | 8);
  gemm(h_ln, wt_v, bv, nullptr, nullptr, h_v, FTOK, DD, DD, 1 | 8);
  vtrans_kernel<<<dim3(TP / 32, HH, BB), 256, 0, stream>>>(h_v, h_vt);
  {
    int qtiles = (TT + 15) / 16;                 // 94
    dim3 g((qtiles + 7) / 8, HH, BB);            // (12, 8, 8)
    attn_wmma_kernel<<<g, 256, 0, stream>>>(h_q, h_k, h_vt, h_o);
  }
  gemm(h_o, wt_o, bo, buf_x, buf_x, nullptr, FTOK, DD, DD, 1 | 4);

  // FFN block
  layernorm_kernel<_Float16><<<FTOK, 256, 0, stream>>>(buf_x, ffn_ln_g, ffn_ln_b, h_ln);
  gemm(h_ln, wt_ff1, ffn_b1, nullptr, nullptr, h_ff, FTOK, FFD, DD, 1 | 2 | 8);
  gemm(h_ff, wt_ff2, ffn_b2, buf_x, buf_x, nullptr, FTOK, DD, FFD, 1 | 4);

  // final LN straight into d_out, then commit loss scalar
  layernorm_kernel<float><<<FTOK, 256, 0, stream>>>(buf_x, ln_post_g, ln_post_b, out);
  loss_final_kernel<<<1, 1, 0, stream>>>(buf_loss, out + (size_t)FTOK * DD);
}
